// LayoutLMv2Layer_6296422056371
// MI455X (gfx1250) — matrix operand order repaired and checked by reading, not compiled, not run on hardware
//
#include <hip/hip_runtime.h>
#include <hip/hip_bf16.h>
#include <math.h>

#define SEQ   2048
#define HID   768
#define NHEAD 12
#define HDIM  64
#define INTER 3072
#define MTOT  4096   // B*S

typedef __attribute__((ext_vector_type(16))) __bf16 v16bf;
typedef __attribute__((ext_vector_type(8)))  __bf16 v8bf;
typedef __attribute__((ext_vector_type(8)))  float  v8f;

union ABfrag { v16bf v; struct { v8bf lo, hi; } p; };

__device__ __forceinline__ __bf16 f2bf(float f) {
  unsigned u = __builtin_bit_cast(unsigned, f);
  unsigned r = u + 0x7fffu + ((u >> 16) & 1u);   // round-to-nearest-even
  unsigned short h = (unsigned short)(r >> 16);
  return __builtin_bit_cast(__bf16, h);
}

__device__ __forceinline__ v8f zero8() {
  v8f z;
#pragma unroll
  for (int i = 0; i < 8; ++i) z[i] = 0.f;
  return z;
}

#define WMMA_BF16(A, B, C) \
  __builtin_amdgcn_wmma_f32_16x16x32_bf16(false, (A), false, (B), (short)0, (C), false, false)

// ---------------------------------------------------------------- fp32 -> bf16
__global__ __launch_bounds__(256)
void cvt_bf(const float* __restrict__ in, __bf16* __restrict__ out, int nElem) {
  int i = blockIdx.x * blockDim.x + threadIdx.x;
  int stride = gridDim.x * blockDim.x;
  for (; i < nElem; i += stride) out[i] = f2bf(in[i]);
}

// ------------------------------------------------------------ generic WMMA GEMM
// Y[M,N] = A[M,K] * W[N,K]^T  ; wave tile 16x64, block = 8 waves -> 128x64
// mode 0: QKV split (bias0=q_bias, bias1=v_bias; ob0=Q(scaled), ob1=K, ob2=V^T)
// mode 1: + bias0 + resid -> outf (fp32)
// mode 2: gelu(acc + bias0)  -> ob0 (bf16)
// mode 3: + bias0 + resid -> outf (fp32)
__global__ __launch_bounds__(256)
void gemm_bf16_wmma(const __bf16* __restrict__ A, const __bf16* __restrict__ W,
                    int N, int K, int mode,
                    const float* __restrict__ bias0, const float* __restrict__ bias1,
                    const float* __restrict__ resid,
                    float* __restrict__ outf,
                    __bf16* __restrict__ ob0, __bf16* __restrict__ ob1,
                    __bf16* __restrict__ ob2)
{
  const int lane = threadIdx.x & 31;
  const int wid  = threadIdx.x >> 5;
  const int g    = lane >> 4;
  const int n    = lane & 15;
  const int m0   = blockIdx.y * 128 + wid * 16;
  const int n0   = blockIdx.x * 64;

  v8f c[4];
#pragma unroll
  for (int i = 0; i < 4; ++i) c[i] = zero8();

  const __bf16* arow = A + (size_t)(m0 + n) * K;
  for (int k = 0; k < K; k += 32) {
    ABfrag a;
    const __bf16* ap = arow + k + 16 * g;
    a.p.lo = *(const v8bf*)(ap);
    a.p.hi = *(const v8bf*)(ap + 8);
#pragma unroll
    for (int nt = 0; nt < 4; ++nt) {
      const __bf16* wp = W + (size_t)(n0 + nt * 16 + n) * K + k + 16 * g;
      v16bf bw = *(const v16bf*)wp;
      c[nt] = WMMA_BF16(a.v, bw, c[nt]);
    }
  }

#pragma unroll
  for (int nt = 0; nt < 4; ++nt) {
#pragma unroll
    for (int r = 0; r < 8; ++r) {
      const int row = m0 + r + 8 * g;
      const int col = n0 + nt * 16 + n;
      float val = c[nt][r];
      if (mode == 0) {
        if (col < HID) {
          ob0[(size_t)row * HID + col] = f2bf((val + bias0[col]) * 0.125f);
        } else if (col < 2 * HID) {
          ob1[(size_t)row * HID + (col - HID)] = f2bf(val);
        } else {
          const int hd = col - 2 * HID;
          const int bb = row >> 11, srow = row & (SEQ - 1);
          ob2[((size_t)bb * HID + hd) * SEQ + srow] = f2bf(val + bias1[hd]);
        }
      } else if (mode == 1 || mode == 3) {
        outf[(size_t)row * N + col] = val + bias0[col] + resid[(size_t)row * N + col];
      } else { // mode 2 : exact GELU
        float t = val + bias0[col];
        float gl = 0.5f * t * (1.0f + erff(t * 0.70710678118654752f));
        ob0[(size_t)row * N + col] = f2bf(gl);
      }
    }
  }
}

// --------------------------------------------------------- flash-style attention
// one wave per (b, head, 16-row q tile); scores = Q K^T + rel + rel2d; mask -> 1e-8
__global__ __launch_bounds__(256)
void attn_wmma(const __bf16* __restrict__ Q, const __bf16* __restrict__ Kb,
               const __bf16* __restrict__ Vt,
               const float* __restrict__ rel, const float* __restrict__ rel2,
               const unsigned char* __restrict__ mask,
               __bf16* __restrict__ ctx)
{
  __shared__ __align__(32) __bf16 lds[8][640];   // per-wave 16 x 32 bf16 (pitch 40)
  const int lane = threadIdx.x & 31;
  const int wid  = threadIdx.x >> 5;
  const int g = lane >> 4, n = lane & 15;
  const int gw = blockIdx.x * 8 + wid;
  const int qt = gw & 127;
  const int h  = (gw >> 7) % NHEAD;
  const int b  = gw / (128 * NHEAD);
  const int q0 = qt * 16;

  const __bf16* Qp = Q  + ((size_t)b * SEQ + q0) * HID + h * HDIM;
  const __bf16* Kp = Kb + (size_t)b * SEQ * HID + h * HDIM;
  const __bf16* Vp = Vt + ((size_t)b * HID + h * HDIM) * SEQ;
  const float*  rp  = rel  + (((size_t)b * NHEAD + h) * SEQ + q0) * SEQ;
  const float*  rp2 = rel2 + (((size_t)b * NHEAD + h) * SEQ + q0) * SEQ;
  const unsigned char* mp = mask + (size_t)b * SEQ;

  ABfrag qa[2];
#pragma unroll
  for (int dc = 0; dc < 2; ++dc) {
    const __bf16* qp = Qp + (size_t)n * HID + dc * 32 + 16 * g;
    qa[dc].p.lo = *(const v8bf*)qp;
    qa[dc].p.hi = *(const v8bf*)(qp + 8);
  }

  v8f acc[4];
#pragma unroll
  for (int i = 0; i < 4; ++i) acc[i] = zero8();
  float mrun[8], lrun[8];
#pragma unroll
  for (int r = 0; r < 8; ++r) { mrun[r] = -1e30f; lrun[r] = 0.f; }

  __bf16* lw = &lds[wid][0];

  for (int kk = 0; kk < SEQ; kk += 32) {
    float sc[2][8];
#pragma unroll
    for (int t = 0; t < 2; ++t) {
      const int kc = kk + 16 * t + n;
      const __bf16* kr = Kp + (size_t)kc * HID;
      v16bf b0 = *(const v16bf*)(kr + 16 * g);
      v16bf b1 = *(const v16bf*)(kr + 32 + 16 * g);
      v8f s = zero8();
      s = WMMA_BF16(qa[0].v, b0, s);
      s = WMMA_BF16(qa[1].v, b1, s);
      const bool mk = mp[kc] != 0;
      const float* r0 = rp  + kc;
      const float* r1 = rp2 + kc;
#pragma unroll
      for (int r = 0; r < 8; ++r) {
        size_t off = (size_t)(r + 8 * g) * SEQ;
        float v = s[r] + r0[off] + r1[off];
        sc[t][r] = mk ? 1e-8f : v;
      }
    }
    // per-row max across the 16 lanes of this group
    float rm[8], rs[8], scl[8];
#pragma unroll
    for (int r = 0; r < 8; ++r) rm[r] = fmaxf(sc[0][r], sc[1][r]);
#pragma unroll
    for (int r = 0; r < 8; ++r)
#pragma unroll
      for (int off = 8; off >= 1; off >>= 1)
        rm[r] = fmaxf(rm[r], __shfl_xor(rm[r], off, 32));
#pragma unroll
    for (int r = 0; r < 8; ++r) {
      float mn = fmaxf(mrun[r], rm[r]);
      scl[r] = __expf(mrun[r] - mn);
      mrun[r] = mn;
      float p0 = __expf(sc[0][r] - mn);
      float p1 = __expf(sc[1][r] - mn);
      sc[0][r] = p0; sc[1][r] = p1;
      rs[r] = p0 + p1;
    }
#pragma unroll
    for (int r = 0; r < 8; ++r)
#pragma unroll
      for (int off = 8; off >= 1; off >>= 1)
        rs[r] += __shfl_xor(rs[r], off, 32);
#pragma unroll
    for (int r = 0; r < 8; ++r) lrun[r] = lrun[r] * scl[r] + rs[r];
#pragma unroll
    for (int j = 0; j < 4; ++j)
#pragma unroll
      for (int r = 0; r < 8; ++r) acc[j][r] *= scl[r];

    // restripe P (C/D layout) -> A-fragment layout via per-wave LDS tile
#pragma unroll
    for (int t = 0; t < 2; ++t)
#pragma unroll
      for (int r = 0; r < 8; ++r)
        lw[(r + 8 * g) * 40 + 16 * t + n] = f2bf(sc[t][r]);
    __builtin_amdgcn_wave_barrier();   // LDS is in-order within a wave
    ABfrag pa;
    pa.p.lo = *(const v8bf*)(lw + n * 40 + 16 * g);
    pa.p.hi = *(const v8bf*)(lw + n * 40 + 16 * g + 8);
    __builtin_amdgcn_wave_barrier();

#pragma unroll
    for (int dt = 0; dt < 4; ++dt) {
      const __bf16* vb = Vp + (size_t)(dt * 16 + n) * SEQ + kk + 16 * g;
      v16bf bv = *(const v16bf*)vb;
      acc[dt] = WMMA_BF16(pa.v, bv, acc[dt]);
    }
  }

#pragma unroll
  for (int r = 0; r < 8; ++r) {
    float inv = 1.0f / lrun[r];
    __bf16* cb = ctx + ((size_t)b * SEQ + q0 + r + 8 * g) * HID + h * HDIM;
#pragma unroll
    for (int dt = 0; dt < 4; ++dt)
      cb[dt * 16 + n] = f2bf(acc[dt][r] * inv);
  }
}

// -------------------------------------------------------------------- layernorm
__global__ __launch_bounds__(256)
void ln768(const float* __restrict__ x, const float* __restrict__ gam,
           const float* __restrict__ bet, float* __restrict__ yf,
           __bf16* __restrict__ yb)
{
  const int row = blockIdx.x, tid = threadIdx.x;
  const float* xr = x + (size_t)row * HID;
  float v[3];
#pragma unroll
  for (int j = 0; j < 3; ++j) v[j] = xr[tid + j * 256];
  __shared__ float red[256];
  red[tid] = v[0] + v[1] + v[2];
  __syncthreads();
  for (int o = 128; o > 0; o >>= 1) { if (tid < o) red[tid] += red[tid + o]; __syncthreads(); }
  const float mean = red[0] * (1.0f / HID);
  __syncthreads();
  float d[3], s2 = 0.f;
#pragma unroll
  for (int j = 0; j < 3; ++j) { d[j] = v[j] - mean; s2 += d[j] * d[j]; }
  red[tid] = s2;
  __syncthreads();
  for (int o = 128; o > 0; o >>= 1) { if (tid < o) red[tid] += red[tid + o]; __syncthreads(); }
  const float rstd = rsqrtf(red[0] * (1.0f / HID) + 1e-12f);
#pragma unroll
  for (int j = 0; j < 3; ++j) {
    const int i = tid + j * 256;
    float y = d[j] * rstd * gam[i] + bet[i];
    yf[(size_t)row * HID + i] = y;
    if (yb) yb[(size_t)row * HID + i] = f2bf(y);
  }
}

// ------------------------------------------------------------------------ launch
extern "C" void kernel_launch(void* const* d_in, const int* in_sizes, int n_in,
                              void* d_out, int out_size, void* d_ws, size_t ws_size,
                              hipStream_t stream) {
  const float* hs    = (const float*)d_in[0];
  const unsigned char* mask = (const unsigned char*)d_in[1];
  const float* rel   = (const float*)d_in[2];
  const float* rel2  = (const float*)d_in[3];
  const float* qkv_w = (const float*)d_in[4];
  const float* q_b   = (const float*)d_in[5];
  const float* v_b   = (const float*)d_in[6];
  const float* aow   = (const float*)d_in[7];
  const float* aob   = (const float*)d_in[8];
  const float* ln1g  = (const float*)d_in[9];
  const float* ln1b  = (const float*)d_in[10];
  const float* iw    = (const float*)d_in[11];
  const float* ib    = (const float*)d_in[12];
  const float* ow    = (const float*)d_in[13];
  const float* ob    = (const float*)d_in[14];
  const float* ln2g  = (const float*)d_in[15];
  const float* ln2b  = (const float*)d_in[16];
  float* out = (float*)d_out;

  char* ws = (char*)d_ws;
  size_t off = 0;
  auto alloc = [&](size_t bytes) { char* p = ws + off; off = (off + bytes + 255) & ~(size_t)255; return p; };

  __bf16* hsb    = (__bf16*)alloc((size_t)MTOT * HID * 2);
  __bf16* qkvwb  = (__bf16*)alloc((size_t)3 * HID * HID * 2);
  __bf16* aowb   = (__bf16*)alloc((size_t)HID * HID * 2);
  __bf16* iwb    = (__bf16*)alloc((size_t)INTER * HID * 2);
  __bf16* owb    = (__bf16*)alloc((size_t)HID * INTER * 2);
  __bf16* Qb     = (__bf16*)alloc((size_t)MTOT * HID * 2);
  __bf16* Kbf    = (__bf16*)alloc((size_t)MTOT * HID * 2);
  __bf16* Vtb    = (__bf16*)alloc((size_t)MTOT * HID * 2);
  __bf16* ctxb   = (__bf16*)alloc((size_t)MTOT * HID * 2);
  float*  res1   = (float*)alloc((size_t)MTOT * HID * 4);
  float*  aof    = (float*)alloc((size_t)MTOT * HID * 4);
  __bf16* aobf   = (__bf16*)alloc((size_t)MTOT * HID * 2);
  __bf16* interb = (__bf16*)alloc((size_t)MTOT * INTER * 2);
  float*  res2   = (float*)alloc((size_t)MTOT * HID * 4);

  // bf16 conversions
  cvt_bf<<<2048, 256, 0, stream>>>(hs,    hsb,   MTOT * HID);
  cvt_bf<<<2048, 256, 0, stream>>>(qkv_w, qkvwb, 3 * HID * HID);
  cvt_bf<<<1024, 256, 0, stream>>>(aow,   aowb,  HID * HID);
  cvt_bf<<<2048, 256, 0, stream>>>(iw,    iwb,   INTER * HID);
  cvt_bf<<<2048, 256, 0, stream>>>(ow,    owb,   HID * INTER);

  // fused QKV projection (mode 0)
  gemm_bf16_wmma<<<dim3(3 * HID / 64, MTOT / 128), 256, 0, stream>>>(
      hsb, qkvwb, 3 * HID, HID, 0, q_b, v_b, nullptr, nullptr, Qb, Kbf, Vtb);

  // flash attention
  attn_wmma<<<384, 256, 0, stream>>>(Qb, Kbf, Vtb, rel, rel2, mask, ctxb);

  // attention output projection + residual (mode 1)
  gemm_bf16_wmma<<<dim3(HID / 64, MTOT / 128), 256, 0, stream>>>(
      ctxb, aowb, HID, HID, 1, aob, nullptr, hs, res1, nullptr, nullptr, nullptr);

  // LN1 -> attn_out (fp32 residual + bf16 for FFN)
  ln768<<<MTOT, 256, 0, stream>>>(res1, ln1g, ln1b, aof, aobf);

  // FFN intermediate with exact GELU (mode 2)
  gemm_bf16_wmma<<<dim3(INTER / 64, MTOT / 128), 256, 0, stream>>>(
      aobf, iwb, INTER, HID, 2, ib, nullptr, nullptr, nullptr, interb, nullptr, nullptr);

  // FFN output + residual (mode 3)
  gemm_bf16_wmma<<<dim3(HID / 64, MTOT / 128), 256, 0, stream>>>(
      interb, owb, HID, INTER, 3, ob, nullptr, aof, res2, nullptr, nullptr, nullptr);

  // LN2 -> final output
  ln768<<<MTOT, 256, 0, stream>>>(res2, ln2g, ln2b, out, nullptr);

  (void)in_sizes; (void)n_in; (void)out_size; (void)ws_size;
}
